// TransitionNER_21492016349738
// MI455X (gfx1250) — matrix operand order repaired and checked by reading, not compiled, not run on hardware
//
#include <hip/hip_runtime.h>

// ---------------- problem constants ----------------
static constexpr int kH  = 1024;   // hidden
static constexpr int kE  = 1024;   // word embedding
static constexpr int kA  = 128;    // action embedding
static constexpr int kL  = 512;    // sequence length
static constexpr int kG  = 4096;   // 4*H gate rows
static constexpr int kNO = 6;      // output classes

static constexpr int NBLK_BUF = 64;   // persistent blocks, buffer scan
static constexpr int NBLK_TRN = 41;   // persistent blocks, transition scan

// ---------------- types / helpers ----------------
typedef __attribute__((ext_vector_type(16))) __bf16 v16bf;
typedef __attribute__((ext_vector_type(8)))  float  v8f;

struct __align__(16) U4 { unsigned x, y, z, w; };
union FragU { U4 q[2]; v16bf v; };

__device__ __forceinline__ float bf2f_lo(unsigned p) {      // low bf16 of a packed pair
  return __builtin_bit_cast(float, p << 16);
}
__device__ __forceinline__ float bf2f_hi(unsigned p) {      // high bf16 of a packed pair
  return __builtin_bit_cast(float, p & 0xFFFF0000u);
}
__device__ __forceinline__ unsigned short f2bf_rne(float f) {
  unsigned u = __builtin_bit_cast(unsigned, f);
  u = u + 0x7FFFu + ((u >> 16) & 1u);
  return (unsigned short)(u >> 16);
}
__device__ __forceinline__ float sigmf(float x) { return 1.0f / (1.0f + __expf(-x)); }

// bf16-weight x f32-vector dot product; 128-bit weight loads, 4 independent
// accumulators so the fma chain has 4-way ILP (fmaf is not reassociable).
__device__ __forceinline__ float dot_bf(const unsigned short* __restrict__ w,
                                        const float* __restrict__ h, int K) {
  float s0 = 0.f, s1 = 0.f, s2 = 0.f, s3 = 0.f;
#pragma unroll 4
  for (int k = 0; k < K; k += 8) {
    U4 q = *(const U4*)(w + k);
    s0 = fmaf(bf2f_lo(q.x), h[k + 0], s0);
    s1 = fmaf(bf2f_hi(q.x), h[k + 1], s1);
    s2 = fmaf(bf2f_lo(q.y), h[k + 2], s2);
    s3 = fmaf(bf2f_hi(q.y), h[k + 3], s3);
    s0 = fmaf(bf2f_lo(q.z), h[k + 4], s0);
    s1 = fmaf(bf2f_hi(q.z), h[k + 5], s1);
    s2 = fmaf(bf2f_lo(q.w), h[k + 6], s2);
    s3 = fmaf(bf2f_hi(q.w), h[k + 7], s3);
  }
  return (s0 + s1) + (s2 + s3);
}

// generation-counter grid barrier for persistent kernels
__device__ __forceinline__ void gridBarrier(unsigned* arrive, unsigned* gen, unsigned nblk) {
  __syncthreads();
  if (threadIdx.x == 0) {
    __threadfence();
    unsigned g = __hip_atomic_load(gen, __ATOMIC_ACQUIRE, __HIP_MEMORY_SCOPE_AGENT);
    unsigned a = __hip_atomic_fetch_add(arrive, 1u, __ATOMIC_ACQ_REL, __HIP_MEMORY_SCOPE_AGENT);
    if (a == nblk - 1u) {
      __hip_atomic_store(arrive, 0u, __ATOMIC_RELAXED, __HIP_MEMORY_SCOPE_AGENT);
      __hip_atomic_fetch_add(gen, 1u, __ATOMIC_ACQ_REL, __HIP_MEMORY_SCOPE_AGENT);
    } else {
      while (__hip_atomic_load(gen, __ATOMIC_ACQUIRE, __HIP_MEMORY_SCOPE_AGENT) == g)
        __builtin_amdgcn_s_sleep(2);
    }
    __threadfence();
  }
  __syncthreads();
}

// ---------------- kernels ----------------

// f32 -> bf16 (RNE) bulk convert
__global__ void k_f2bf(const float* __restrict__ src, unsigned short* __restrict__ dst, int n) {
  for (int i = blockIdx.x * blockDim.x + threadIdx.x; i < n; i += gridDim.x * blockDim.x)
    dst[i] = f2bf_rne(src[i]);
}

// gather word embeddings for the sentence, emit bf16
__global__ void k_gather_tok(const int* __restrict__ sent, const float* __restrict__ wemb,
                             unsigned short* __restrict__ tok_bf) {
  int t = blockIdx.x;
  const float* src = wemb + (size_t)sent[t] * kE;
  for (int i = threadIdx.x; i < kE; i += blockDim.x)
    tok_bf[(size_t)t * kE + i] = f2bf_rne(src[i]);
}

// gather action embeddings, emit bf16
__global__ void k_gather_act(const int* __restrict__ acts, const float* __restrict__ aemb,
                             unsigned short* __restrict__ out_bf) {
  int t = blockIdx.x;
  const float* src = aemb + (size_t)acts[t] * kA;
  for (int i = threadIdx.x; i < kA; i += blockDim.x)
    out_bf[(size_t)t * kA + i] = f2bf_rne(src[i]);
}

// ec[r] = lin1_b[r] + dot(lin1_W[r, H:2H], empty_emb)   (constant empty-NER slot)
__global__ void k_ec(const float* __restrict__ lin1W, const float* __restrict__ lin1b,
                     const float* __restrict__ empty, float* __restrict__ ec) {
  int r = blockIdx.x * blockDim.x + threadIdx.x;
  if (r < kH) {
    const float* w = lin1W + (size_t)r * kG + kH;
    float s0 = lin1b[r], s1 = 0.f, s2 = 0.f, s3 = 0.f;
    for (int k = 0; k < kH; k += 4) {
      s0 = fmaf(w[k + 0], empty[k + 0], s0);
      s1 = fmaf(w[k + 1], empty[k + 1], s1);
      s2 = fmaf(w[k + 2], empty[k + 2], s2);
      s3 = fmaf(w[k + 3], empty[k + 3], s3);
    }
    ec[r] = (s0 + s1) + (s2 + s3);
  }
}

// init persistent state + barriers
__global__ void k_init(const float* __restrict__ h0, const float* __restrict__ c0,
                       const float* __restrict__ empty,
                       float* buf_h, float* bufC,
                       float* out_h, float* out_c, float* act_h, float* act_c,
                       float* featOut, float* featAct, unsigned* bars) {
  for (int j = threadIdx.x; j < kH; j += blockDim.x) {
    float h = h0[j], c = c0[j], e = empty[j];
    buf_h[j] = h; bufC[j] = c;
    out_h[j] = h; out_c[j] = c; act_h[j] = h; act_c[j] = c;
    featOut[j] = e; featAct[j] = e;
  }
  if (threadIdx.x < 64) bars[threadIdx.x] = 0u;
}

// C[m][n] = sum_k A[m][k]*B[n][k] + bias0[n] + bias1[n]
// A:[M,lda] bf16 row-major, B:[N,ldb] bf16 row-major, C:[M,ldc] f32.
// One wave per 16x16 tile, K-loop of 32 via v_wmma_f32_16x16x32_bf16.
__global__ void k_wmma_gemm_bt(const unsigned short* __restrict__ Abf, int lda,
                               const unsigned short* __restrict__ Bbf, int ldb,
                               float* __restrict__ C, int ldc,
                               const float* __restrict__ bias0,
                               const float* __restrict__ bias1,
                               int M, int N, int K) {
  const int lane = threadIdx.x & 31;
  const int wave = threadIdx.x >> 5;
  const int tile = blockIdx.x * (blockDim.x >> 5) + wave;
  const int ntn  = N >> 4;
  const int mt   = tile / ntn;
  const int nt   = tile - mt * ntn;
  if (mt >= (M >> 4)) return;                    // wave-uniform: EXEC stays all-ones
  const int  lm = lane & 15;
  const bool hi = (lane & 16) != 0;

  // A 16x32 bf16 frag: lanes 0-15 hold K {0..7,16..23}; lanes 16-31 hold K {8..15,24..31}
  const unsigned short* arow = Abf + (size_t)(mt * 16 + lm) * lda + (hi ? 8 : 0);
  // B 32x16 frag: lanes 0-15 hold K {0..7,16..23}; lanes 16-31 hold K {8..15,24..31}
  const unsigned short* brow = Bbf + (size_t)(nt * 16 + lm) * ldb + (hi ? 8 : 0);

  v8f acc = {0.f, 0.f, 0.f, 0.f, 0.f, 0.f, 0.f, 0.f};
  for (int k0 = 0; k0 < K; k0 += 32) {
    FragU fa, fb;
    fa.q[0] = *(const U4*)(arow + k0);
    fa.q[1] = *(const U4*)(arow + k0 + 16);
    fb.q[0] = *(const U4*)(brow + k0);
    fb.q[1] = *(const U4*)(brow + k0 + 16);
    acc = __builtin_amdgcn_wmma_f32_16x16x32_bf16(false, fa.v, false, fb.v,
                                                  (short)0, acc, false, false);
  }
  const int n = nt * 16 + lm;
  float bv = 0.f;
  if (bias0) bv += bias0[n];
  if (bias1) bv += bias1[n];
  const int mbase = mt * 16 + (hi ? 8 : 0);      // D: lanes 0-15 -> M=v, 16-31 -> M=v+8
#pragma unroll
  for (int v = 0; v < 8; ++v)
    C[(size_t)(mbase + v) * ldc + n] = acc[v] + bv;
}

// Buffer LSTM scan (tokens in reverse). Persistent, 64 blocks.
// 4 lanes per gate row (K split 4x256), gates -> global, grid sync, pointwise.
__global__ void k_buf_scan(const float* __restrict__ IGbuf,          // [L][4H] x-gates (rev-indexed)
                           const unsigned short* __restrict__ Whh,   // [4H][H] bf16
                           float* __restrict__ buf_h,                // [L+1][H]
                           float* __restrict__ cstate,               // [H]
                           float* __restrict__ gates,                // [4H]
                           unsigned* __restrict__ bar) {
  const int tid  = blockIdx.x * 256 + threadIdx.x;   // 0..16383
  const int row  = tid >> 2;                         // 0..4095
  const int ksub = tid & 3;                          // 256-wide K slice
  __shared__ float hsh[kH];
  const unsigned short* wrow = Whh + (size_t)row * kH + ksub * 256;

  for (int t = 0; t < kL; ++t) {
    const float* hsrc = buf_h + (size_t)t * kH;
    for (int i = threadIdx.x; i < kH; i += 256) hsh[i] = hsrc[i];
    __syncthreads();

    // prefetch next step's input-gate row while this GEMV runs (global_prefetch_b8)
    if (t + 1 < kL && ksub == 0)
      __builtin_prefetch(IGbuf + (size_t)(kL - 2 - t) * kG + row, 0, 1);

    float s = dot_bf(wrow, hsh + ksub * 256, 256);
    s += __shfl_xor(s, 1, 32);
    s += __shfl_xor(s, 2, 32);
    if (ksub == 0) gates[row] = s;

    gridBarrier(bar, bar + 1, NBLK_BUF);

    if (tid < kH) {
      const float* ig = IGbuf + (size_t)(kL - 1 - t) * kG;  // reversed token order
      float gi = ig[tid]            + gates[tid];
      float gf = ig[kH + tid]       + gates[kH + tid];
      float gg = ig[2 * kH + tid]   + gates[2 * kH + tid];
      float go = ig[3 * kH + tid]   + gates[3 * kH + tid];
      float c  = sigmf(gf) * cstate[tid] + sigmf(gi) * tanhf(gg);
      cstate[tid] = c;
      buf_h[(size_t)(t + 1) * kH + tid] = sigmf(go) * tanhf(c);
    }
    gridBarrier(bar, bar + 1, NBLK_BUF);
  }
}

// bufTop_bf[t] = bf16(buf_h[L - t])   (top-of-buffer feature per step)
__global__ void k_buftop(const float* __restrict__ buf_h, unsigned short* __restrict__ top_bf) {
  int t = blockIdx.x;
  const float* src = buf_h + (size_t)(kL - t) * kH;
  for (int i = threadIdx.x; i < kH; i += blockDim.x)
    top_bf[(size_t)t * kH + i] = f2bf_rne(src[i]);
}

// Transition scan. Persistent, 41 blocks x 256:
//  stage A: recurrent GEMVs (out:4096 rows, act:4096 rows, lin1 recurrent: 2048 half-rows)
//           + block 40 computes logits of step t-1 (overlapped)
//  stage B: LSTM pointwise updates + hid = tanh(pre + dLin)
__global__ void k_trans_scan(const float* __restrict__ IGout, const float* __restrict__ IGact,
                             const float* __restrict__ pre,
                             const unsigned short* __restrict__ Wout,
                             const unsigned short* __restrict__ Wact,
                             const unsigned short* __restrict__ W1bf,   // [H][4H] bf16
                             const float* __restrict__ lin2W, const float* __restrict__ lin2b,
                             float* out_h, float* out_c, float* act_h, float* act_c,
                             float* featOut, float* featAct,
                             float* dOut, float* dAct, float* dLin, float* hid,
                             unsigned* __restrict__ bar, float* __restrict__ logits) {
  const int tid = blockIdx.x * 256 + threadIdx.x;
  __shared__ float sh_oh[kH], sh_ah[kH], sh_fo[kH], sh_fa[kH];

  for (int t = 0; t < kL; ++t) {
    for (int i = threadIdx.x; i < kH; i += 256) {
      sh_oh[i] = out_h[i];  sh_ah[i] = act_h[i];
      sh_fo[i] = featOut[i]; sh_fa[i] = featAct[i];
    }
    __syncthreads();

    if (tid < 4096) {
      if (t + 1 < kL) __builtin_prefetch(IGout + (size_t)(t + 1) * kG + tid, 0, 1);
      dOut[tid] = dot_bf(Wout + (size_t)tid * kH, sh_oh, kH);
    } else if (tid < 8192) {
      int r = tid - 4096;
      if (t + 1 < kL) __builtin_prefetch(IGact + (size_t)(t + 1) * kG + r, 0, 1);
      dAct[r] = dot_bf(Wact + (size_t)r * kH, sh_ah, kH);
    } else if (tid < 10240) {
      int u = tid - 8192, j = u >> 1, half = u & 1;
      if (t + 1 < kL && half == 0) __builtin_prefetch(pre + (size_t)(t + 1) * kH + j, 0, 1);
      const unsigned short* w = W1bf + (size_t)j * kG + (half ? 3 * kH : 2 * kH);
      float s = dot_bf(w, half ? sh_fa : sh_fo, kH);
      s += __shfl_xor(s, 1, 32);
      if (half == 0) dLin[j] = s;
    } else if (t > 0) {
      int local = tid - 10240;                 // block 40: logits of step t-1
      if (local < kNO * 32) {
        int rowo = local >> 5, lane = local & 31;
        float s = 0.f;
        for (int k = lane; k < kH; k += 32) s = fmaf(lin2W[rowo * kH + k], hid[k], s);
        for (int off = 16; off > 0; off >>= 1) s += __shfl_xor(s, off, 32);
        if (lane == 0) logits[(t - 1) * kNO + rowo] = s + lin2b[rowo];
      }
    }

    gridBarrier(bar, bar + 1, NBLK_TRN);

    if (tid < kH) {                                  // out-stack LSTM update
      int j = tid;
      const float* ig = IGout + (size_t)t * kG;
      float gi = ig[j]           + dOut[j];
      float gf = ig[kH + j]      + dOut[kH + j];
      float gg = ig[2 * kH + j]  + dOut[2 * kH + j];
      float go = ig[3 * kH + j]  + dOut[3 * kH + j];
      float c  = sigmf(gf) * out_c[j] + sigmf(gi) * tanhf(gg);
      float h2 = sigmf(go) * tanhf(c);
      out_c[j] = c; out_h[j] = h2; featOut[j] = h2;
    } else if (tid < 2 * kH) {                       // action-stack LSTM update
      int j = tid - kH;
      const float* ig = IGact + (size_t)t * kG;
      float gi = ig[j]           + dAct[j];
      float gf = ig[kH + j]      + dAct[kH + j];
      float gg = ig[2 * kH + j]  + dAct[2 * kH + j];
      float go = ig[3 * kH + j]  + dAct[3 * kH + j];
      float c  = sigmf(gf) * act_c[j] + sigmf(gi) * tanhf(gg);
      float h2 = sigmf(go) * tanhf(c);
      act_c[j] = c; act_h[j] = h2; featAct[j] = h2;
    } else if (tid < 3 * kH) {                       // hid = tanh(pre + recurrent lin1 part)
      int j = tid - 2 * kH;
      hid[j] = tanhf(pre[(size_t)t * kH + j] + dLin[j]);
    }

    gridBarrier(bar, bar + 1, NBLK_TRN);
  }

  // final logits (t = L-1)
  if (tid >= 10240) {
    int local = tid - 10240;
    if (local < kNO * 32) {
      int rowo = local >> 5, lane = local & 31;
      float s = 0.f;
      for (int k = lane; k < kH; k += 32) s = fmaf(lin2W[rowo * kH + k], hid[k], s);
      for (int off = 16; off > 0; off >>= 1) s += __shfl_xor(s, off, 32);
      if (lane == 0) logits[(kL - 1) * kNO + rowo] = s + lin2b[rowo];
    }
  }
}

// ---------------- host launcher ----------------
extern "C" void kernel_launch(void* const* d_in, const int* in_sizes, int n_in,
                              void* d_out, int out_size, void* d_ws, size_t ws_size,
                              hipStream_t stream) {
  (void)in_sizes; (void)n_in; (void)out_size;
  const int*   sent   = (const int*)d_in[0];
  const int*   acts   = (const int*)d_in[1];
  const float* wemb   = (const float*)d_in[2];
  const float* aembT  = (const float*)d_in[3];
  const float* bufWih = (const float*)d_in[4];
  const float* bufWhh = (const float*)d_in[5];
  const float* bufbih = (const float*)d_in[6];
  const float* bufbhh = (const float*)d_in[7];
  const float* outWih = (const float*)d_in[8];
  const float* outWhh = (const float*)d_in[9];
  const float* outbih = (const float*)d_in[10];
  const float* outbhh = (const float*)d_in[11];
  const float* actWih = (const float*)d_in[12];
  const float* actWhh = (const float*)d_in[13];
  const float* actbih = (const float*)d_in[14];
  const float* actbhh = (const float*)d_in[15];
  const float* empty  = (const float*)d_in[16];
  const float* lin1W  = (const float*)d_in[17];
  const float* lin1b  = (const float*)d_in[18];
  const float* lin2W  = (const float*)d_in[19];
  const float* lin2b  = (const float*)d_in[20];
  const float* h0     = (const float*)d_in[21];
  const float* c0     = (const float*)d_in[22];
  float* logits = (float*)d_out;

  char* base = (char*)d_ws;
  size_t off = 0;
  auto carve = [&](size_t bytes) -> void* {
    void* p = base + off;
    off += (bytes + 255) & ~(size_t)255;
    return p;
  };

  // bf16 weight copies (L2-resident working set ~49 MB)
  unsigned short* bufWih_bf = (unsigned short*)carve((size_t)kG * kE * 2);
  unsigned short* bufWhh_bf = (unsigned short*)carve((size_t)kG * kH * 2);
  unsigned short* outWih_bf = (unsigned short*)carve((size_t)kG * kE * 2);
  unsigned short* outWhh_bf = (unsigned short*)carve((size_t)kG * kH * 2);
  unsigned short* actWih_bf = (unsigned short*)carve((size_t)kG * kA * 2);
  unsigned short* actWhh_bf = (unsigned short*)carve((size_t)kG * kH * 2);
  unsigned short* lin1W_bf  = (unsigned short*)carve((size_t)kH * kG * 2);
  // activations
  unsigned short* tok_bf    = (unsigned short*)carve((size_t)kL * kE * 2);
  unsigned short* aemb_bf   = (unsigned short*)carve((size_t)kL * kA * 2);
  unsigned short* buftop_bf = (unsigned short*)carve((size_t)kL * kH * 2);
  float* IGbuf  = (float*)carve((size_t)kL * kG * 4);
  float* IGout  = (float*)carve((size_t)kL * kG * 4);
  float* IGact  = (float*)carve((size_t)kL * kG * 4);
  float* pre    = (float*)carve((size_t)kL * kH * 4);
  float* ec     = (float*)carve((size_t)kH * 4);
  float* buf_h  = (float*)carve((size_t)(kL + 1) * kH * 4);
  float* bufC   = (float*)carve((size_t)kH * 4);
  float* bufGat = (float*)carve((size_t)kG * 4);
  float* out_h  = (float*)carve((size_t)kH * 4);
  float* out_c  = (float*)carve((size_t)kH * 4);
  float* act_h  = (float*)carve((size_t)kH * 4);
  float* act_c  = (float*)carve((size_t)kH * 4);
  float* featO  = (float*)carve((size_t)kH * 4);
  float* featA  = (float*)carve((size_t)kH * 4);
  float* dOut   = (float*)carve((size_t)kG * 4);
  float* dAct   = (float*)carve((size_t)kG * 4);
  float* dLin   = (float*)carve((size_t)kH * 4);
  float* hid    = (float*)carve((size_t)kH * 4);
  unsigned* bars = (unsigned*)carve(64 * sizeof(unsigned));
  if (off > ws_size) return;  // workspace too small; ~82 MB needed

  // 0) state + barrier init
  k_init<<<1, 256, 0, stream>>>(h0, c0, empty, buf_h, bufC, out_h, out_c, act_h, act_c,
                                featO, featA, bars);

  // 1) weight conversion to bf16
  k_f2bf<<<1024, 256, 0, stream>>>(bufWih, bufWih_bf, kG * kE);
  k_f2bf<<<1024, 256, 0, stream>>>(bufWhh, bufWhh_bf, kG * kH);
  k_f2bf<<<1024, 256, 0, stream>>>(outWih, outWih_bf, kG * kE);
  k_f2bf<<<1024, 256, 0, stream>>>(outWhh, outWhh_bf, kG * kH);
  k_f2bf<<<1024, 256, 0, stream>>>(actWih, actWih_bf, kG * kA);
  k_f2bf<<<1024, 256, 0, stream>>>(actWhh, actWhh_bf, kG * kH);
  k_f2bf<<<1024, 256, 0, stream>>>(lin1W,  lin1W_bf,  kH * kG);

  // 2) embedding gathers + empty-slot constant
  k_gather_tok<<<kL, 256, 0, stream>>>(sent, wemb, tok_bf);
  k_gather_act<<<kL, 128, 0, stream>>>(acts, aembT, aemb_bf);
  k_ec<<<4, 256, 0, stream>>>(lin1W, lin1b, empty, ec);

  // 3) input-side gate GEMMs (WMMA bf16, batched over all 512 steps)
  //    tiles = (512/16)*(4096/16) = 8192; 8 waves/block -> 1024 blocks
  k_wmma_gemm_bt<<<1024, 256, 0, stream>>>(tok_bf, kE, bufWih_bf, kE, IGbuf, kG,
                                           bufbih, bufbhh, kL, kG, kE);
  k_wmma_gemm_bt<<<1024, 256, 0, stream>>>(tok_bf, kE, outWih_bf, kE, IGout, kG,
                                           outbih, outbhh, kL, kG, kE);
  k_wmma_gemm_bt<<<1024, 256, 0, stream>>>(aemb_bf, kA, actWih_bf, kA, IGact, kG,
                                           actbih, actbhh, kL, kG, kA);

  // 4) sequential buffer LSTM scan (persistent + grid barriers)
  k_buf_scan<<<NBLK_BUF, 256, 0, stream>>>(IGbuf, bufWhh_bf, buf_h, bufC, bufGat, bars);

  // 5) lin1 precompute over buffer tops: pre[t] = bufTop(t)@W1[:, :H]^T + ec
  k_buftop<<<kL, 256, 0, stream>>>(buf_h, buftop_bf);
  k_wmma_gemm_bt<<<256, 256, 0, stream>>>(buftop_bf, kH, lin1W_bf, kG, pre, kH,
                                          ec, nullptr, kL, kH, kH);

  // 6) sequential transition scan (out/act LSTMs + classifier)
  k_trans_scan<<<NBLK_TRN, 256, 0, stream>>>(IGout, IGact, pre, outWhh_bf, actWhh_bf,
                                             lin1W_bf, lin2W, lin2b,
                                             out_h, out_c, act_h, act_c, featO, featA,
                                             dOut, dAct, dLin, hid, bars + 32, logits);
}
